// IntLinear_56882546868738
// MI455X (gfx1250) — 8-bit matrix operand order repaired and checked by reading, compiled, not run on hardware
//
#include <hip/hip_runtime.h>

typedef __attribute__((ext_vector_type(8))) int v8i;

#define M_DIM 16384
#define K_DIM 1024
#define N_DIM 1024
#define SCALE_INT 10
#define OUT_ZP   (-5)
#define IN_ZP    3

__global__ void pack_i8(const int* __restrict__ in, signed char* __restrict__ out, int n4) {
    int i = blockIdx.x * blockDim.x + threadIdx.x;
    if (i >= n4) return;
    int4 v = reinterpret_cast<const int4*>(in)[i];
    unsigned packed = (unsigned(v.x) & 0xffu)
                    | ((unsigned(v.y) & 0xffu) << 8)
                    | ((unsigned(v.z) & 0xffu) << 16)
                    | (unsigned(v.w) << 24);
    reinterpret_cast<unsigned*>(out)[i] = packed;
}

__global__ void corr_kernel(const int* __restrict__ w32, const int* __restrict__ bias,
                            int* __restrict__ corr) {
    int n = blockIdx.x;
    const int* row = w32 + (size_t)n * K_DIM;
    int s = 0;
    for (int k = threadIdx.x; k < K_DIM; k += blockDim.x) s += row[k];
    __shared__ int red[256];
    red[threadIdx.x] = s;
    __syncthreads();
    for (int off = 128; off > 0; off >>= 1) {
        if ((int)threadIdx.x < off) red[threadIdx.x] += red[threadIdx.x + off];
        __syncthreads();
    }
    if (threadIdx.x == 0) corr[n] = bias[n] - IN_ZP * red[0];
}

__device__ __forceinline__ v8i load_a_frag(const signed char* p) {
    int2 q0 = *reinterpret_cast<const int2*>(p);
    int2 q1 = *reinterpret_cast<const int2*>(p + 16);
    int2 q2 = *reinterpret_cast<const int2*>(p + 32);
    int2 q3 = *reinterpret_cast<const int2*>(p + 48);
    v8i a;
    a[0] = q0.x; a[1] = q0.y; a[2] = q1.x; a[3] = q1.y;
    a[4] = q2.x; a[5] = q2.y; a[6] = q3.x; a[7] = q3.y;
    return a;
}

__device__ __forceinline__ v8i load_b_frag(const signed char* p) {
    int2 q0 = *reinterpret_cast<const int2*>(p);
    int2 q1 = *reinterpret_cast<const int2*>(p + 16);
    int2 q2 = *reinterpret_cast<const int2*>(p + 32);
    int2 q3 = *reinterpret_cast<const int2*>(p + 48);
    v8i b;
    b[0] = q0.x; b[1] = q0.y; b[2] = q1.x; b[3] = q1.y;
    b[4] = q2.x; b[5] = q2.y; b[6] = q3.x; b[7] = q3.y;
    return b;
}

__global__ __launch_bounds__(256) void gemm_iu8(const signed char* __restrict__ A8,
                                                const signed char* __restrict__ B8,
                                                const int* __restrict__ corr,
                                                float* __restrict__ out) {
    const int lane = threadIdx.x & 31;
    const int wave = threadIdx.x >> 5;
    const int r    = lane & 15;
    const int half = lane >> 4;
    const int wm   = wave >> 1;
    const int wn   = wave & 1;
    const int waveM = blockIdx.y * 128 + wm * 32;
    const int waveN = blockIdx.x * 128 + wn * 64;

    v8i acc[2][4];
    const v8i vzero = {0, 0, 0, 0, 0, 0, 0, 0};
#pragma unroll
    for (int mi = 0; mi < 2; ++mi)
#pragma unroll
        for (int ni = 0; ni < 4; ++ni) acc[mi][ni] = vzero;

    const signed char* aBase[2];
#pragma unroll
    for (int mi = 0; mi < 2; ++mi)
        aBase[mi] = A8 + (size_t)(waveM + mi * 16 + r) * K_DIM + half * 8;

    const signed char* bBase[4];
#pragma unroll
    for (int ni = 0; ni < 4; ++ni)
        bBase[ni] = B8 + (size_t)(waveN + ni * 16 + r) * K_DIM + half * 8;

    const int colLane = waveN + r;
    const int cvReg[4] = { corr[colLane], corr[colLane + 16],
                           corr[colLane + 32], corr[colLane + 48] };
    float* outBase = out + (size_t)(waveM + half * 8) * N_DIM + colLane;

    for (int kb = 0; kb < K_DIM; kb += 64) {
        v8i a[2];
#pragma unroll
        for (int mi = 0; mi < 2; ++mi) a[mi] = load_a_frag(aBase[mi] + kb);
        v8i b[4];
#pragma unroll
        for (int ni = 0; ni < 4; ++ni) b[ni] = load_b_frag(bBase[ni] + kb);

#pragma unroll
        for (int mi = 0; mi < 2; ++mi)
#pragma unroll
            for (int ni = 0; ni < 4; ++ni)
                acc[mi][ni] = __builtin_amdgcn_wmma_i32_16x16x64_iu8(
                     true, a[mi],
                     true, b[ni],
                    acc[mi][ni],
                     false,  false);
    }

#pragma unroll
    for (int ni = 0; ni < 4; ++ni) {
        const int cv = cvReg[ni];
#pragma unroll
        for (int mi = 0; mi < 2; ++mi) {
            float* op = outBase + (size_t)(mi * 16) * N_DIM + ni * 16;
#pragma unroll
            for (int j = 0; j < 8; ++j) {
                int v = acc[mi][ni][j] + cv;
                int t = (v * SCALE_INT) >> 10;
                t += OUT_ZP;
                t = t < -128 ? -128 : (t > 127 ? 127 : t);
                op[(size_t)j * N_DIM] = (float)t;
            }
        }
    }
}

extern "C" void kernel_launch(void* const* d_in, const int* in_sizes, int n_in,
                              void* d_out, int out_size, void* d_ws, size_t ws_size,
                              hipStream_t stream) {
    const int* x32  = (const int*)d_in[0];
    const int* w32  = (const int*)d_in[1];
    const int* bias = (const int*)d_in[2];
    float* out = (float*)d_out;

    signed char* x8 = (signed char*)d_ws;
    signed char* w8 = x8 + (size_t)M_DIM * K_DIM;
    int* corr = (int*)(w8 + (size_t)N_DIM * K_DIM);

    const int n4x = (M_DIM * K_DIM) / 4;
    pack_i8<<<(n4x + 255) / 256, 256, 0, stream>>>(x32, x8, n4x);
    const int n4w = (N_DIM * K_DIM) / 4;
    pack_i8<<<(n4w + 255) / 256, 256, 0, stream>>>(w32, w8, n4w);

    corr_kernel<<<N_DIM, 256, 0, stream>>>(w32, bias, corr);

    dim3 grid(N_DIM / 128, M_DIM / 128);
    gemm_iu8<<<grid, 256, 0, stream>>>(x8, w8, corr, out);
}
